// MultiHeadAttentionLayer_5858335391780
// MI455X (gfx1250) — matrix operand order repaired and checked by reading, not compiled, not run on hardware
//
#include <hip/hip_runtime.h>
#include <hip/hip_bf16.h>

// MI455X / gfx1250: wave32, WMMA bf16 path (v_wmma_f32_16x16x32_bf16),
// async-to-LDS staging (ASYNCcnt) for the flash-attention K/V tiles.
// MHA: B=4, S=2048, D=1024, H=16, DH=64.

#define DEVI static __device__ __forceinline__

typedef __attribute__((ext_vector_type(16))) __bf16 v16bf;
typedef __attribute__((ext_vector_type(8)))  float  v8f;

union FragU { uint4 q[2]; v16bf v; };

DEVI int lane_id() { return (int)(threadIdx.x & 31u); }

DEVI v8f vzero8() {
  v8f r = {0.f, 0.f, 0.f, 0.f, 0.f, 0.f, 0.f, 0.f};
  return r;
}

DEVI v8f wmma_bf16(v16bf a, v16bf b, v8f c) {
  // D = A(16x32 bf16) * B(32x16 bf16) + C(16x16 f32)
  return __builtin_amdgcn_wmma_f32_16x16x32_bf16(
      /*neg_a=*/false, a, /*neg_b=*/false, b,
      /*c_mod=*/(short)0, c, /*reuse_a=*/false, /*reuse_b=*/false);
}

// A-fragment (16x32, bf16) from row-major storage; row_base points at row 0 of the tile.
// Lane l: M = l&15, half = l>>4. VGPR v<4: K = half*8+2v ; v>=4: K = 16+half*8+2(v-4).
DEVI v16bf load_a16(const __bf16* row_base, int lda, int kofs) {
  const int l = lane_id();
  const int m = l & 15, hf = l >> 4;
  const __bf16* p = row_base + (size_t)m * lda + (kofs + hf * 8);
  FragU f;
  f.q[0] = *(const uint4*)(p);
  f.q[1] = *(const uint4*)(p + 16);
  return f.v;
}

// B-fragment (32x16, bf16) from "column-contiguous" storage T[n][k] (k contiguous).
// Lane l: N = l&15, half = l>>4. VGPR v<4: K = half*8+2v ; v>=4: K = 16+half*8+2(v-4).
DEVI v16bf load_b16(const __bf16* col_base, int ldt, int kofs) {
  const int l = lane_id();
  const int n = l & 15, hf = l >> 4;
  const __bf16* p = col_base + (size_t)n * ldt + (kofs + hf * 8);
  FragU f;
  f.q[0] = *(const uint4*)(p);
  f.q[1] = *(const uint4*)(p + 16);
  return f.v;
}

// A-fragment from a 16x32 row-major bf16 LDS tile (P-matrix layout swap).
DEVI v16bf load_a16_lds(const __bf16* tile) {
  const int l = lane_id();
  const int m = l & 15, hf = l >> 4;
  const __bf16* p = tile + m * 32 + hf * 8;
  FragU f;
  f.q[0] = *(const uint4*)(p);
  f.q[1] = *(const uint4*)(p + 16);
  return f.v;
}

// Async 16B global -> LDS copy (gfx1250, tracked by ASYNCcnt).
// VDST VGPR = workgroup-relative LDS byte address = low 32 bits of the generic
// pointer (ISA 10.2: LDS aperture maps addr[31:0] to the wave's LDS space).
DEVI void cp_async_b128(const __bf16* g, const __bf16* l) {
  unsigned int lofs = (unsigned int)(unsigned long long)l;
  asm volatile("global_load_async_to_lds_b128 %0, %1, off"
               :: "v"(lofs), "v"(g) : "memory");
}

DEVI void wait_async_le(int prev_outstanding) {
  if (prev_outstanding > 0)
    asm volatile("s_wait_asynccnt 0x4" ::: "memory");
  else
    asm volatile("s_wait_asynccnt 0x0" ::: "memory");
}

// ---------------------------------------------------------------------------
// Kernel 0a: x f32 -> bf16 (8 elements / thread, b128 in, b128 out)
// ---------------------------------------------------------------------------
__global__ void mha_cvt_x_bf16(const float* __restrict__ x,
                               __bf16* __restrict__ xh, int n8) {
  int i = blockIdx.x * blockDim.x + threadIdx.x;
  if (i >= n8) return;
  const float4* p = (const float4*)x + (size_t)i * 2;
  float4 a = p[0], c = p[1];
  union { __bf16 h[8]; uint4 u; } o;
  o.h[0] = (__bf16)a.x; o.h[1] = (__bf16)a.y;
  o.h[2] = (__bf16)a.z; o.h[3] = (__bf16)a.w;
  o.h[4] = (__bf16)c.x; o.h[5] = (__bf16)c.y;
  o.h[6] = (__bf16)c.z; o.h[7] = (__bf16)c.w;
  ((uint4*)xh)[i] = o.u;
}

// ---------------------------------------------------------------------------
// Kernel 0b: W[k][n] f32 -> WT[n][k] bf16 for the 3 weight matrices
// ---------------------------------------------------------------------------
__global__ void mha_transpose_w(const float* __restrict__ Wq,
                                const float* __restrict__ Wk,
                                const float* __restrict__ Wv,
                                __bf16* __restrict__ WT) {
  const int z = blockIdx.z;
  const float* W = (z == 0) ? Wq : (z == 1) ? Wk : Wv;
  __bf16* T = WT + (size_t)z * 1024 * 1024;
  int idx = blockIdx.x * 256 + threadIdx.x;      // 0 .. 1048575
  int n = idx & 1023, k = idx >> 10;
  T[(size_t)n * 1024 + k] = (__bf16)W[(size_t)k * 1024 + n];
}

// ---------------------------------------------------------------------------
// Kernel 1: QKV projection GEMM, [8192x1024]*[1024x1024]+bias, WMMA bf16.
// Software-pipelined: next k-step's A + 4 B fragments are loaded into fresh
// registers while the current 4 WMMAs issue (partial loadcnt waits).
// ---------------------------------------------------------------------------
__global__ __launch_bounds__(128)
void mha_qkv_gemm(const __bf16* __restrict__ xh, const __bf16* __restrict__ WT,
                  const float* __restrict__ bq, const float* __restrict__ bk,
                  const float* __restrict__ bv,
                  __bf16* __restrict__ Qh, __bf16* __restrict__ Kh,
                  __bf16* __restrict__ VTv) {
  const int z = blockIdx.z;
  const __bf16* Wz = WT + (size_t)z * 1024 * 1024;
  const int m0 = blockIdx.x * 16;
  const int wave = threadIdx.x >> 5;
  const int n0 = blockIdx.y * 256 + wave * 64;

  v8f acc0 = vzero8(), acc1 = vzero8(), acc2 = vzero8(), acc3 = vzero8();
  const __bf16* arow = xh + (size_t)m0 * 1024;
  const __bf16* w0 = Wz + (size_t)(n0 +  0) * 1024;
  const __bf16* w1 = Wz + (size_t)(n0 + 16) * 1024;
  const __bf16* w2 = Wz + (size_t)(n0 + 32) * 1024;
  const __bf16* w3 = Wz + (size_t)(n0 + 48) * 1024;

  v16bf a_c = load_a16(arow, 1024, 0);
  v16bf b0  = load_b16(w0, 1024, 0);
  v16bf b1  = load_b16(w1, 1024, 0);
  v16bf b2  = load_b16(w2, 1024, 0);
  v16bf b3  = load_b16(w3, 1024, 0);

  for (int k = 32; k < 1024; k += 32) {
    v16bf a_n = load_a16(arow, 1024, k);
    v16bf c0  = load_b16(w0, 1024, k);
    v16bf c1  = load_b16(w1, 1024, k);
    v16bf c2  = load_b16(w2, 1024, k);
    v16bf c3  = load_b16(w3, 1024, k);
    acc0 = wmma_bf16(a_c, b0, acc0);
    acc1 = wmma_bf16(a_c, b1, acc1);
    acc2 = wmma_bf16(a_c, b2, acc2);
    acc3 = wmma_bf16(a_c, b3, acc3);
    a_c = a_n; b0 = c0; b1 = c1; b2 = c2; b3 = c3;
  }
  acc0 = wmma_bf16(a_c, b0, acc0);
  acc1 = wmma_bf16(a_c, b1, acc1);
  acc2 = wmma_bf16(a_c, b2, acc2);
  acc3 = wmma_bf16(a_c, b3, acc3);

  const float* bias = (z == 0) ? bq : (z == 1) ? bk : bv;
  const float qscale = (z == 0) ? 0.125f : 1.0f;   // fold 1/sqrt(DH) into Q
  const int l = lane_id(), nn = l & 15, hf = l >> 4;

  v8f accs[4] = {acc0, acc1, acc2, acc3};
#pragma unroll
  for (int j = 0; j < 4; ++j) {
    const int col = n0 + j * 16 + nn;
    const int h = col >> 6, d = col & 63;
    const float bcol = bias[col];
#pragma unroll
    for (int r = 0; r < 8; ++r) {
      const int row = m0 + r + hf * 8;
      const int bb = row >> 11, s = row & 2047;
      const float val = (accs[j][r] + bcol) * qscale;
      if (z == 2)
        VTv[(((size_t)(bb * 16 + h)) * 64 + d) * 2048 + s] = (__bf16)val;
      else if (z == 0)
        Qh[(((size_t)(bb * 16 + h)) * 2048 + s) * 64 + d] = (__bf16)val;
      else
        Kh[(((size_t)(bb * 16 + h)) * 2048 + s) * 64 + d] = (__bf16)val;
    }
  }
}

// ---------------------------------------------------------------------------
// Kernel 2: flash attention. grid (B*H=64, S/64=32), block 128 = 4 waves.
// K/V tiles (shared by all 4 waves) are staged into double-buffered LDS with
// global_load_async_to_lds_b128 (ASYNCcnt): issue tile i+1, wait asynccnt<=4
// so tile i (in-order per wave) is complete, barrier, compute 8 WMMAs.
// ---------------------------------------------------------------------------
__global__ __launch_bounds__(128)
void mha_flash_attn(const __bf16* __restrict__ Qh, const __bf16* __restrict__ Kh,
                    const __bf16* __restrict__ VTv, const float* __restrict__ mask,
                    float* __restrict__ out) {
  __shared__ __bf16 KtS[2][32 * 64];   // 2 x 4KB: 32 keys x 64 d (row-major)
  __shared__ __bf16 VtS[2][64 * 32];   // 2 x 4KB: 64 dcol x 32 k (k contiguous)
  __shared__ __bf16 Pt[4][16 * 32];    // per-wave P tile (C->A layout swap)

  const int head = blockIdx.x;                 // b*16 + h
  const int b = head >> 4, h = head & 15;
  const int tid = threadIdx.x;
  const int wave = tid >> 5;
  const int q0 = (blockIdx.y * 4 + wave) * 16;

  const __bf16* Qb = Qh + ((size_t)head * 2048 + q0) * 64;
  const __bf16* Kb = Kh + (size_t)head * 2048 * 64;
  const __bf16* Vb = VTv + (size_t)head * 64 * 2048;
  const float*  mk = mask + (size_t)b * 2048;

  const int l = lane_id(), nn = l & 15, hf = l >> 4;

  const v16bf qa0 = load_a16(Qb, 64, 0);   // Q tile, d = 0..31  (pre-scaled by 1/8)
  const v16bf qa1 = load_a16(Qb, 64, 32);  // Q tile, d = 32..63

  float mrun[8], lrun[8];
#pragma unroll
  for (int r = 0; r < 8; ++r) { mrun[r] = -3.0e38f; lrun[r] = 0.0f; }
  v8f O0 = vzero8(), O1 = vzero8(), O2 = vzero8(), O3 = vzero8();

  // Cooperative async stage of one 32-key tile into buffer bi:
  // K tile is 4KB contiguous (rows k0..k0+31 of [S,DH]); V tile is 64 rows of
  // 64B taken from [DH,S] at column offset k0. 4 x b128 per thread.
  auto stage = [&](int k0, int bi) {
    const __bf16* gk = Kb + (size_t)k0 * 64 + tid * 8;
    cp_async_b128(gk,        &KtS[bi][tid * 8]);
    cp_async_b128(gk + 1024, &KtS[bi][tid * 8 + 1024]);
    const int c0 = tid, c1 = tid + 128;            // 256 chunks of 16B
    cp_async_b128(Vb + (size_t)(c0 >> 2) * 2048 + k0 + (c0 & 3) * 8,
                  &VtS[bi][(c0 >> 2) * 32 + (c0 & 3) * 8]);
    cp_async_b128(Vb + (size_t)(c1 >> 2) * 2048 + k0 + (c1 & 3) * 8,
                  &VtS[bi][(c1 >> 2) * 32 + (c1 & 3) * 8]);
  };

  stage(0, 0);

  for (int it = 0; it < 64; ++it) {
    const int k0 = it * 32;
    const int bi = it & 1;
    const bool more = (it + 1) < 64;
    if (more) stage(k0 + 32, bi ^ 1);      // overlap next tile with this compute
    wait_async_le(more ? 4 : 0);           // tile `it` landed (in-order per wave)
    __syncthreads();                       // visible to all 4 waves

    const __bf16* Kc = KtS[bi];
    const __bf16* Vc = VtS[bi];

    // ---- scores: two 16x16 tiles over 32 keys, accumulate over DH=64 ----
    v8f s0 = vzero8(), s1 = vzero8();
    s0 = wmma_bf16(qa0, load_b16(Kc,            64, 0),  s0);
    s0 = wmma_bf16(qa1, load_b16(Kc,            64, 32), s0);
    s1 = wmma_bf16(qa0, load_b16(Kc + 16 * 64,  64, 0),  s1);
    s1 = wmma_bf16(qa1, load_b16(Kc + 16 * 64,  64, 32), s1);

    // additive mask penalty: column = key index = same for all 8 rows in lane
    const float pen0 = -10000.0f * (1.0f - mk[k0 + nn]);
    const float pen1 = -10000.0f * (1.0f - mk[k0 + 16 + nn]);

    __bf16* tile = Pt[wave];
#pragma unroll
    for (int r = 0; r < 8; ++r) {
      const float s0r = s0[r] + pen0;
      const float s1r = s1[r] + pen1;
      float tmax = fmaxf(s0r, s1r);        // row reduce: xor<=8 stays in-half
#pragma unroll
      for (int sh = 8; sh >= 1; sh >>= 1)
        tmax = fmaxf(tmax, __shfl_xor(tmax, sh, 32));
      const float mnew  = fmaxf(mrun[r], tmax);
      const float alpha = __expf(mrun[r] - mnew);
      const float p0 = __expf(s0r - mnew);
      const float p1 = __expf(s1r - mnew);
      float ls = p0 + p1;
#pragma unroll
      for (int sh = 8; sh >= 1; sh >>= 1)
        ls += __shfl_xor(ls, sh, 32);
      lrun[r] = lrun[r] * alpha + ls;
      mrun[r] = mnew;
      O0[r] *= alpha; O1[r] *= alpha; O2[r] *= alpha; O3[r] *= alpha;
      const int row = r + hf * 8;          // C-layout -> LDS (16x32 row-major)
      tile[row * 32 + nn]      = (__bf16)p0;
      tile[row * 32 + 16 + nn] = (__bf16)p1;
    }
    __syncthreads();                       // P tiles ready

    // ---- P(16x32) * V(32x16) into 4 output column tiles (DH=64) ----
    const v16bf pa = load_a16_lds(tile);
    O0 = wmma_bf16(pa, load_b16(Vc +  0 * 32, 32, 0), O0);
    O1 = wmma_bf16(pa, load_b16(Vc + 16 * 32, 32, 0), O1);
    O2 = wmma_bf16(pa, load_b16(Vc + 32 * 32, 32, 0), O2);
    O3 = wmma_bf16(pa, load_b16(Vc + 48 * 32, 32, 0), O3);
    __syncthreads();                       // buffer `bi` free for reuse at it+2
  }

  // ---- epilogue: normalize by l and scatter f32 to [B,S,D] ----
#pragma unroll
  for (int r = 0; r < 8; ++r) {
    const float inv = 1.0f / lrun[r];
    const int row = q0 + r + hf * 8;
    float* op = out + ((size_t)b * 2048 + row) * 1024 + h * 64 + nn;
    op[0]  = O0[r] * inv;
    op[16] = O1[r] * inv;
    op[32] = O2[r] * inv;
    op[48] = O3[r] * inv;
  }
}

// ---------------------------------------------------------------------------
// Launch
// ---------------------------------------------------------------------------
extern "C" void kernel_launch(void* const* d_in, const int* in_sizes, int n_in,
                              void* d_out, int out_size, void* d_ws, size_t ws_size,
                              hipStream_t stream) {
  const float* x    = (const float*)d_in[0];
  const float* mask = (const float*)d_in[1];
  const float* Wq   = (const float*)d_in[2];
  const float* bq   = (const float*)d_in[3];
  const float* Wk   = (const float*)d_in[4];
  const float* bk   = (const float*)d_in[5];
  const float* Wv   = (const float*)d_in[6];
  const float* bv   = (const float*)d_in[7];
  float* out = (float*)d_out;

  char* ws = (char*)d_ws;
  __bf16* xh = (__bf16*)(ws);                         // 16 MiB: x in bf16 [8192,1024]
  __bf16* WT = (__bf16*)(ws + (16ull << 20));         //  6 MiB: 3x [1024(out),1024(in)] bf16
  __bf16* Qh = (__bf16*)(ws + (22ull << 20));         // 16 MiB: [B,H,S,DH] bf16 (x 1/8)
  __bf16* Kh = (__bf16*)(ws + (38ull << 20));         // 16 MiB: [B,H,S,DH] bf16
  __bf16* VT = (__bf16*)(ws + (54ull << 20));         // 16 MiB: [B,H,DH,S] bf16  (end 70 MiB)

  mha_cvt_x_bf16<<<4096, 256, 0, stream>>>(x, xh, (8192 * 1024) / 8);
  mha_transpose_w<<<dim3(4096, 1, 3), 256, 0, stream>>>(Wq, Wk, Wv, WT);
  mha_qkv_gemm<<<dim3(512, 4, 3), 128, 0, stream>>>(xh, WT, bq, bk, bv, Qh, Kh, VT);
  mha_flash_attn<<<dim3(64, 32), 128, 0, stream>>>(Qh, Kh, VT, mask, out);
}
